// LMDA_38551626449124
// MI455X (gfx1250) — compile-verified, matrix operand order checked by reading, not run on hardware
//
#include <hip/hip_runtime.h>
#include <math.h>

// ---------------- model constants ----------------
#define BB     256
#define CH     22
#define SAMP   1125
#define CD1    24
#define CD2    9
#define KT     75
#define W1     1051            // SAMP - KT + 1
#define WP     210             // W1 / 5
#define NN     (BB * W1)       // 269056 rows for Cheb GEMM
#define KP     224             // padded K for Cheb (198 logical = 9*22)
#define NP     224             // padded N for Cheb
#define NTILE  14              // NP/16
#define KSTEPS 7               // KP/32
#define WELEM  (KP * NP)       // 50176 halfs per layer weight
#define SH_CHEB (WELEM * 2)    // 100352 B dynamic LDS for the W tile

typedef _Float16 h16;
typedef __attribute__((ext_vector_type(16))) _Float16 v16h;
typedef __attribute__((ext_vector_type(8)))  float    v8f;
typedef __attribute__((ext_vector_type(4)))  unsigned int u32x4;
typedef __attribute__((ext_vector_type(4)))  int      i32x4;
typedef __attribute__((ext_vector_type(8)))  int      i32x8;

// ---------------- workspace layout ----------------
constexpr size_t alignup(size_t x) { return (x + 255) & ~(size_t)255; }
constexpr size_t OFF_G     = 0;                                            // 24*22 f32
constexpr size_t OFF_Q     = alignup(OFF_G     + 24*22*4);                 // 24 f32
constexpr size_t OFF_S3T3  = alignup(OFF_Q     + 24*4);                    // 18 f32
constexpr size_t OFF_S4T4  = alignup(OFF_S3T3  + 18*4);                    // 18 f32
constexpr size_t OFF_APAD  = alignup(OFF_S4T4  + 18*4);                    // 32*96 f16
constexpr size_t OFF_ADJ   = alignup(OFF_APAD  + 32*96*2);                 // 3*484 f32
constexpr size_t OFF_SCR   = alignup(OFF_ADJ   + 3*484*4);                 // 2*484 f32
constexpr size_t OFF_MSUM  = alignup(OFF_SCR   + 2*484*4);                 // 5*50176 f16 (fragment-swizzled)
constexpr size_t OFF_BIASM = alignup(OFF_MSUM  + (size_t)5*WELEM*2);       // 5*224 f32
constexpr size_t OFF_H     = alignup(OFF_BIASM + 5*NP*4);                  // B*24*22*W1 f16
constexpr size_t OFF_HM    = alignup(OFF_H     + (size_t)BB*CD1*CH*W1*2);  // B*24*W1 f32
constexpr size_t OFF_ATTN  = alignup(OFF_HM    + (size_t)BB*CD1*W1*4);     // B*24*W1 f32
constexpr size_t OFF_XA    = alignup(OFF_ATTN  + (size_t)BB*CD1*W1*4);     // NN*KP f16
constexpr size_t OFF_XB    = alignup(OFF_XA    + (size_t)NN*KP*2);         // NN*KP f16
constexpr size_t OFF_POOL  = alignup(OFF_XB    + (size_t)NN*KP*2);         // B*1890 f32

// ---------------- helpers ----------------
__device__ __forceinline__ float geluf(float x) {
    return 0.5f * x * (1.f + erff(x * 0.70710678118654752f));
}
__device__ __forceinline__ float sigmf(float x) { return 1.f / (1.f + expf(-x)); }

// 16-bit A-matrix (16x32) per-lane K mapping (ISA 7.12.2); B uses the mirrored map.
__device__ __forceinline__ int kmap(int grp, int j) {
    int p = j >> 1, lo = j & 1;
    int base = (p < 4) ? (2 * p + lo) : (16 + 2 * (p - 4) + lo);
    return base + grp * 8;
}

// ---------------- prep: fold BN/1x1, build padded WMMA operands, Laplacian, Msum ----------------
__global__ void k_prep(const float* cw, const float* tc1, const float* bn1,
                       const float* tc2, const float* bn2, const float* bn3,
                       const float* bn4, const float* A, const float* Amask,
                       const float* chW, const float* chB,
                       float* G, float* Q, float* s3t3, float* s4t4,
                       h16* apad, float* adj, float* scr, h16* msum, float* biasm) {
    int t = threadIdx.x;
    __shared__ float eff[528], ss1[24], st1[24], ss2[24], st2[24], sT2[24], sdis[22];

    if (t < 528) {
        int o = t / 22, c = t % 22;
        float a = 0.f;
        for (int i = 0; i < 9; ++i) a += tc1[o * 9 + i] * cw[i * 22 + c];
        eff[t] = a;
    }
    if (t < 24) {
        float g = bn1[t], b = bn1[24 + t], m = bn1[48 + t], v = bn1[72 + t];
        float s = g * rsqrtf(v + 1e-5f); ss1[t] = s; st1[t] = b - m * s;
        g = bn2[t]; b = bn2[24 + t]; m = bn2[48 + t]; v = bn2[72 + t];
        s = g * rsqrtf(v + 1e-5f); ss2[t] = s; st2[t] = b - m * s;
        float T = 0.f;
        for (int k = 0; k < KT; ++k) T += tc2[t * KT + k];
        sT2[t] = T;
    }
    __syncthreads();
    if (t < 528) { int o = t / 22; G[t] = ss2[o] * ss1[o] * eff[t]; }
    if (t < 24)  Q[t] = ss2[t] * st1[t] * sT2[t] + st2[t];
    if (t < 9) {
        float g = bn3[t], b = bn3[9 + t], m = bn3[18 + t], v = bn3[27 + t];
        float s = g * rsqrtf(v + 1e-5f); s3t3[t] = s; s3t3[9 + t] = b - m * s;
        g = bn4[t]; b = bn4[9 + t]; m = bn4[18 + t]; v = bn4[27 + t];
        s = g * rsqrtf(v + 1e-5f); s4t4[t] = s; s4t4[9 + t] = b - m * s;
    }
    for (int i = t; i < 32 * 96; i += 256) {
        int m = i / 96, k = i % 96;
        apad[i] = (m < CD1 && k < KT) ? (h16)tc2[m * KT + k] : (h16)0.f;
    }
    // Laplacian:  Lhat = -D^{-1/2} (relu(A*sig(M)) symmetrized, zero-diag) D^{-1/2}
    float* tmpA = scr; float* S = scr + 484;
    for (int i = t; i < 484; i += 256) {
        int c = i / 22, d = i % 22;
        float v = A[i] * sigmf(Amask[i]);
        v = fmaxf(v, 0.f); if (c == d) v = 0.f;
        tmpA[i] = v;
    }
    __syncthreads();
    for (int i = t; i < 484; i += 256) { int c = i / 22, d = i % 22; S[i] = tmpA[i] + tmpA[d * 22 + c]; }
    __syncthreads();
    if (t < 22) {
        float ds = 0.f;
        for (int d = 0; d < 22; ++d) ds += S[t * 22 + d];
        sdis[t] = rsqrtf(ds + 1e-10f);
    }
    __syncthreads();
    for (int i = t; i < 484; i += 256) {
        int c = i / 22, d = i % 22;
        adj[484 + i] = -sdis[c] * S[i] * sdis[d];
        adj[i] = (c == d) ? 1.f : 0.f;
    }
    __syncthreads();
    for (int i = t; i < 484; i += 256) {
        int c = i / 22, d = i % 22;
        float a2 = 0.f;
        for (int e = 0; e < 22; ++e) a2 += adj[484 + c * 22 + e] * adj[484 + e * 22 + d];
        adj[968 + i] = 2.f * a2 - ((c == d) ? 1.f : 0.f);
    }
    __syncthreads();
    // Msum[(f,d)][(o,c)] = sum_k W[l,k][f,o]*adj_k[c,d], stored FRAGMENT-SWIZZLED:
    //   msum[l][kc][nt][lane][j]  (kc<7, nt<14, lane<32, j<16) so each lane's
    //   WMMA B-fragment is one contiguous 32-byte run in LDS.
    for (int idx = t; idx < 5 * WELEM; idx += 256) {
        int l = idx / WELEM, rem = idx % WELEM;
        int kc = rem / 7168, r2 = rem % 7168;       // 7168 = 14*32*16
        int nt = r2 / 512,  r3 = r2 % 512;
        int lane = r3 / 16, j = r3 % 16;
        int m = lane & 15, grp = lane >> 4;
        int r = kc * 32 + kmap(grp, j);             // K index (f*22+d)
        int i = nt * 16 + m;                        // N index (o*22+c)
        float v = 0.f;
        if (r < 198 && i < 198) {
            int f = r / 22, d = r % 22, o = i / 22, c = i % 22;
            for (int k = 0; k < 3; ++k)
                v += chW[((l * 3 + k) * 9 + f) * 9 + o] * adj[k * 484 + c * 22 + d];
        }
        msum[idx] = (h16)v;
    }
    for (int idx = t; idx < 5 * NP; idx += 256) {
        int l = idx / NP, i = idx % NP;
        float v = 0.f;
        if (i < 198) { int o = i / 22; for (int k = 0; k < 3; ++k) v += chB[(l * 3 + k) * 9 + o]; }
        biasm[idx] = v;
    }
}

// ---------------- stage 1: depthwise (1,75) time conv as im2col WMMA + folded BN+GELU ----------------
__global__ void k_tconv(const float* x, const h16* apad, const float* G, const float* Q, h16* h) {
    int b = blockIdx.x / CH, c = blockIdx.x % CH;
    int t = threadIdx.x, wave = t >> 5, lane = t & 31;
    int m = lane & 15, grp = lane >> 4;

    __shared__ h16 sxr[1152];
    __shared__ h16 sA[32 * 96];
    for (int i = t; i < 1152; i += 256)
        sxr[i] = (i < SAMP) ? (h16)x[((size_t)b * CH + c) * SAMP + i] : (h16)0.f;
    for (int i = t; i < 32 * 96; i += 256) sA[i] = apad[i];
    __syncthreads();

    v16h af[2][3];
#pragma unroll
    for (int mt = 0; mt < 2; ++mt)
#pragma unroll
        for (int kc = 0; kc < 3; ++kc)
#pragma unroll
            for (int j = 0; j < 16; ++j)
                af[mt][kc][j] = sA[(mt * 16 + m) * 96 + kc * 32 + kmap(grp, j)];

    const v8f z8 = {0.f, 0.f, 0.f, 0.f, 0.f, 0.f, 0.f, 0.f};
    for (int nt = wave; nt < 66; nt += 8) {
        int w0 = nt * 16;
        v8f a0 = z8, a1 = z8;
#pragma unroll
        for (int kc = 0; kc < 3; ++kc) {
            v16h bf;
#pragma unroll
            for (int j = 0; j < 16; ++j)
                bf[j] = sxr[w0 + m + kc * 32 + kmap(grp, j)];
            a0 = __builtin_amdgcn_wmma_f32_16x16x32_f16(false, af[0][kc], false, bf, (short)0, a0, false, false);
            a1 = __builtin_amdgcn_wmma_f32_16x16x32_f16(false, af[1][kc], false, bf, (short)0, a1, false, false);
        }
        int w = w0 + m;
        if (w < W1) {
            int hi = grp * 8;
#pragma unroll
            for (int r = 0; r < 8; ++r) {
                int o = r + hi;                 // rows 0..15 of A-tile 0
                float hv = geluf(G[o * 22 + c] * a0[r] + Q[o]);
                h[(((size_t)b * CD1 + o) * CH + c) * W1 + w] = (h16)hv;
                int o2 = 16 + r + hi;           // rows 16..31 of A-tile 1 (24..31 padded)
                if (o2 < CD1) {
                    float hv2 = geluf(G[o2 * 22 + c] * a1[r] + Q[o2]);
                    h[(((size_t)b * CD1 + o2) * CH + c) * W1 + w] = (h16)hv2;
                }
            }
        }
    }
}

// ---------------- stage 2: channel mean for depth attention ----------------
__global__ void k_mean(const h16* h, float* hm) {
    int idx = blockIdx.x * 256 + threadIdx.x;
    if (idx >= BB * CD1 * W1) return;
    int w = idx % W1, o = (idx / W1) % CD1, b = idx / (W1 * CD1);
    float s = 0.f;
    for (int c = 0; c < CH; ++c)
        s += (float)h[(((size_t)b * CD1 + o) * CH + c) * W1 + w];
    hm[idx] = s * (1.f / 22.f);
}

// ---------------- stage 3: 7-tap depth conv + softmax over depth ----------------
__global__ void k_attn(const float* hm, const float* daw, const float* dab, float* attn) {
    int idx = blockIdx.x * 256 + threadIdx.x;
    if (idx >= BB * W1) return;
    int b = idx / W1, w = idx % W1;
    float p[24], q[24];
#pragma unroll
    for (int j = 0; j < 24; ++j) p[j] = hm[((size_t)b * CD1 + j) * W1 + w];
    float bias = dab[0], mx = -1e30f;
#pragma unroll
    for (int j = 0; j < 24; ++j) {
        float a = bias;
#pragma unroll
        for (int tt = 0; tt < 7; ++tt) {
            int jj = j + tt - 3;
            if (jj >= 0 && jj < 24) a += daw[tt] * p[jj];
        }
        q[j] = a; mx = fmaxf(mx, a);
    }
    float sum = 0.f;
#pragma unroll
    for (int j = 0; j < 24; ++j) { q[j] = expf(q[j] - mx); sum += q[j]; }
    float inv = 1.f / sum;
#pragma unroll
    for (int j = 0; j < 24; ++j) attn[((size_t)b * CD1 + j) * W1 + w] = q[j] * inv;
}

// ---------------- stage 4: attention-scale + 1x1 cconv1 + BN3 -> Cheb input X0 ----------------
__global__ void k_cconv1(const h16* h, const float* attn, const float* cc1w,
                         const float* s3t3, h16* X0) {
    __shared__ float scw[216];
    __shared__ float sst[18];
    int t = threadIdx.x;
    for (int i = t; i < 216; i += 256) scw[i] = cc1w[i];
    if (t < 18) sst[t] = s3t3[t];
    __syncthreads();

    int idx = blockIdx.x * 256 + t;               // exactly NN*22 threads
    int n = idx / CH, c = idx % CH;
    int b = n / W1, w = n % W1;
    float acc[9] = {0.f, 0.f, 0.f, 0.f, 0.f, 0.f, 0.f, 0.f, 0.f};
    for (int i = 0; i < CD1; ++i) {
        float av = attn[((size_t)b * CD1 + i) * W1 + w];
        float hv = (float)h[(((size_t)b * CD1 + i) * CH + c) * W1 + w];
        float val = 22.f * av * hv;
#pragma unroll
        for (int f = 0; f < 9; ++f) acc[f] += scw[f * 24 + i] * val;
    }
    h16* row = X0 + (size_t)n * KP;
#pragma unroll
    for (int f = 0; f < 9; ++f) row[f * 22 + c] = (h16)(sst[f] * acc[f] + sst[9 + f]);
    if (c == 0)
        for (int p = 198; p < KP; ++p) row[p] = (h16)0.f;
}

// ---------------- stage 5: Cheb layer GEMM: Y = relu(X*Msum + bias) ----------------
// W tile (100KB, fragment-swizzled) lands in LDS via one TDM tensor_load_to_lds;
// A fragments stream straight from global (coalesced b128), no per-k barriers.
__global__ void __launch_bounds__(256) k_cheb(const h16* __restrict__ X, h16* __restrict__ Y,
                                              const h16* __restrict__ Wl, const float* __restrict__ biasl) {
    extern __shared__ h16 sW[];                   // WELEM halfs
    int t = threadIdx.x, wave = t >> 5, lane = t & 31;
    int m = lane & 15, grp = lane >> 4;
    size_t m0 = (size_t)blockIdx.x * 128;
    int wr0 = wave * 16;

#if __has_builtin(__builtin_amdgcn_tensor_load_to_lds)
    if (wave == 0) {
        // D# group0: count=1 | lds_addr | 57-bit global addr | type=2
        unsigned long long ga = (unsigned long long)(size_t)Wl;
        u32x4 g0;
        g0[0] = 1u;
        g0[1] = (unsigned)(size_t)sW;
        g0[2] = (unsigned)(ga & 0xFFFFFFFFull);
        g0[3] = (unsigned)((ga >> 32) & 0x01FFFFFFull) | (2u << 30);
        // D# group1: 1-D copy of WELEM 2-byte elements (tile_dim0=50176 fits u16)
        i32x8 g1;
        g1[0] = (int)(1u << 16);                  // data_size = 2B
        g1[1] = (int)((WELEM & 0xFFFF) << 16);    // tensor_dim0[15:0] @ [63:48]
        g1[2] = (int)(((WELEM >> 16) & 0xFFFF) | (1u << 16)); // dim0 hi | tensor_dim1=1
        g1[3] = (int)((WELEM & 0xFFFF) << 16);    // tile_dim0 @ [127:112]
        g1[4] = 0;                                // tile_dim1/2 unused
        g1[5] = (int)WELEM;                       // tensor_dim0_stride low
        g1[6] = 0; g1[7] = 0;
        i32x4 z4 = {0, 0, 0, 0};
#if defined(__clang_major__) && __clang_major__ >= 23
        i32x8 z8i = {0, 0, 0, 0, 0, 0, 0, 0};
        __builtin_amdgcn_tensor_load_to_lds(g0, g1, z4, z4, z8i, 0);
#else
        __builtin_amdgcn_tensor_load_to_lds(g0, g1, z4, z4, 0);
#endif
        __builtin_amdgcn_s_wait_tensorcnt(0);
    }
#else
    for (int i = t; i < WELEM / 8; i += 256)
        ((uint4*)sW)[i] = ((const uint4*)Wl)[i];
#endif
    __syncthreads();

    const v8f z8 = {0.f, 0.f, 0.f, 0.f, 0.f, 0.f, 0.f, 0.f};
    v8f acc[NTILE];
#pragma unroll
    for (int i = 0; i < NTILE; ++i) acc[i] = z8;

    const h16* arow = X + (m0 + wr0 + m) * KP;    // this lane's A row
    for (int kc = 0; kc < KSTEPS; ++kc) {
        // A fragment: two aligned 16B global loads (disjoint across grp)
        union { uint4 q[2]; v16h v; } ua;
        ua.q[0] = *(const uint4*)(arow + kc * 32 + grp * 8);
        ua.q[1] = *(const uint4*)(arow + kc * 32 + 16 + grp * 8);
        v16h a = ua.v;
        if (kc < KSTEPS - 1)
            __builtin_prefetch(arow + (kc + 1) * 32, 0, 1);   // global_prefetch_b8
        const h16* wbase = sW + (size_t)kc * 7168 + lane * 16;
#pragma unroll
        for (int nt = 0; nt < NTILE; ++nt) {
            v16h bf = *(const v16h*)(wbase + nt * 512);       // contiguous 32B -> ds_load_b128 x2
            acc[nt] = __builtin_amdgcn_wmma_f32_16x16x32_f16(false, a, false, bf, (short)0, acc[nt], false, false);
        }
    }
    // epilogue: bias + relu, store f16 (pad cols come out as exact zeros)
#pragma unroll
    for (int nt = 0; nt < NTILE; ++nt) {
        int col = nt * 16 + m;
        float bcol = biasl[col];
#pragma unroll
        for (int r = 0; r < 8; ++r) {
            int rl = wr0 + r + grp * 8;
            float v = fmaxf(acc[nt][r] + bcol, 0.f);
            Y[(m0 + rl) * KP + col] = (h16)v;
        }
    }
}

// ---------------- stage 6: depthwise channel reduce + BN4 + GELU + avgpool5 ----------------
__global__ void k_post1(const h16* Xf, const float* cc2w, const float* s4t4, float* pooled) {
    int idx = blockIdx.x * 256 + threadIdx.x;    // exactly B*9*210
    int wp = idx % WP, f = (idx / WP) % CD2, b = idx / (WP * CD2);
    float s4 = s4t4[f], t4 = s4t4[9 + f];
    float s = 0.f;
    for (int q = 0; q < 5; ++q) {
        int w = wp * 5 + q;
        const h16* row = Xf + ((size_t)b * W1 + w) * KP + f * 22;
        float v = 0.f;
        for (int c = 0; c < CH; ++c) v += (float)row[c] * cc2w[f * 22 + c];
        s += geluf(s4 * v + t4);
    }
    pooled[(size_t)b * 1890 + f * WP + wp] = s * 0.2f;
}

// ---------------- stage 7: fc1 + MLP + softmax ----------------
__global__ void k_head(const float* pooled, const float* fc1w, const float* fc1b,
                       const float* w1, const float* b1, const float* w2, const float* b2,
                       float* out) {
    int b = blockIdx.x, t = threadIdx.x;
    __shared__ float red[128];
    __shared__ float l5[5];
    for (int m = 0; m < 5; ++m) {
        float loc = 0.f;
        for (int j = t; j < 1890; j += 128)
            loc += pooled[(size_t)b * 1890 + j] * fc1w[j * 5 + m];
        red[t] = loc; __syncthreads();
        for (int s = 64; s > 0; s >>= 1) { if (t < s) red[t] += red[t + s]; __syncthreads(); }
        if (t == 0) l5[m] = red[0] + fc1b[m];
        __syncthreads();
    }
    if (t == 0) {
        float h64[64];
        for (int j = 0; j < 64; ++j) {
            float a = b1[j];
            for (int i = 0; i < 5; ++i) a += l5[i] * w1[i * 64 + j];
            h64[j] = (a > 0.f) ? a : (expf(a) - 1.f);    // ELU
        }
        float o4[4], mx = -1e30f;
        for (int m = 0; m < 4; ++m) {
            float a = b2[m];
            for (int j = 0; j < 64; ++j) a += h64[j] * w2[j * 4 + m];
            o4[m] = a; mx = fmaxf(mx, a);
        }
        float sum = 0.f;
        for (int m = 0; m < 4; ++m) { o4[m] = expf(o4[m] - mx); sum += o4[m]; }
        float inv = 1.f / sum;
        for (int m = 0; m < 4; ++m) out[b * 4 + m] = o4[m] * inv;
    }
}

// ---------------- launch ----------------
extern "C" void kernel_launch(void* const* d_in, const int* in_sizes, int n_in,
                              void* d_out, int out_size, void* d_ws, size_t ws_size,
                              hipStream_t stream) {
    const float* x     = (const float*)d_in[0];
    const float* cw    = (const float*)d_in[1];
    const float* tc1   = (const float*)d_in[2];
    const float* bn1   = (const float*)d_in[3];
    const float* tc2   = (const float*)d_in[4];
    const float* bn2   = (const float*)d_in[5];
    const float* daw   = (const float*)d_in[6];
    const float* dab   = (const float*)d_in[7];
    const float* cc1w  = (const float*)d_in[8];
    const float* bn3   = (const float*)d_in[9];
    const float* Amat  = (const float*)d_in[10];
    const float* Amask = (const float*)d_in[11];
    const float* chW   = (const float*)d_in[12];
    const float* chB   = (const float*)d_in[13];
    const float* cc2w  = (const float*)d_in[14];
    const float* bn4   = (const float*)d_in[15];
    const float* fc1w  = (const float*)d_in[16];
    const float* fc1b  = (const float*)d_in[17];
    const float* mw1   = (const float*)d_in[18];
    const float* mb1   = (const float*)d_in[19];
    const float* mw2   = (const float*)d_in[20];
    const float* mb2   = (const float*)d_in[21];
    float* out = (float*)d_out;

    char* w = (char*)d_ws;
    float* G     = (float*)(w + OFF_G);
    float* Q     = (float*)(w + OFF_Q);
    float* s3t3  = (float*)(w + OFF_S3T3);
    float* s4t4  = (float*)(w + OFF_S4T4);
    h16*   apad  = (h16*)  (w + OFF_APAD);
    float* adj   = (float*)(w + OFF_ADJ);
    float* scr   = (float*)(w + OFF_SCR);
    h16*   msum  = (h16*)  (w + OFF_MSUM);
    float* biasm = (float*)(w + OFF_BIASM);
    h16*   hbuf  = (h16*)  (w + OFF_H);
    float* hm    = (float*)(w + OFF_HM);
    float* attn  = (float*)(w + OFF_ATTN);
    h16*   Xa    = (h16*)  (w + OFF_XA);
    h16*   Xb    = (h16*)  (w + OFF_XB);
    float* pool  = (float*)(w + OFF_POOL);

    k_prep<<<1, 256, 0, stream>>>(cw, tc1, bn1, tc2, bn2, bn3, bn4, Amat, Amask,
                                  chW, chB, G, Q, s3t3, s4t4, apad, adj, scr, msum, biasm);
    k_tconv<<<BB * CH, 256, 0, stream>>>(x, apad, G, Q, hbuf);
    k_mean<<<(BB * CD1 * W1) / 256, 256, 0, stream>>>(hbuf, hm);
    k_attn<<<(BB * W1 + 255) / 256, 256, 0, stream>>>(hm, daw, dab, attn);
    k_cconv1<<<(NN * CH) / 256, 256, 0, stream>>>(hbuf, attn, cc1w, s3t3, Xa);

    h16* cur = Xa; h16* nxt = Xb;
    for (int l = 0; l < 5; ++l) {
        k_cheb<<<NN / 128, 256, SH_CHEB, stream>>>(cur, nxt, msum + (size_t)l * WELEM, biasm + l * NP);
        h16* tmp = cur; cur = nxt; nxt = tmp;
    }
    // after 5 layers, result is in `cur`
    k_post1<<<(BB * CD2 * WP) / 256, 256, 0, stream>>>(cur, cc2w, s4t4, pool);
    k_head<<<BB, 128, 0, stream>>>(pool, fc1w, fc1b, mw1, mb1, mw2, mb2, out);
}
